// TBConv2d_1597727834341
// MI455X (gfx1250) — matrix operand order repaired and checked by reading, not compiled, not run on hardware
//
#include <hip/hip_runtime.h>

typedef __attribute__((ext_vector_type(16))) _Float16 v16h;
typedef __attribute__((ext_vector_type(8)))  _Float16 v8h;
typedef __attribute__((ext_vector_type(8)))  float    v8f;
typedef __attribute__((ext_vector_type(4)))  float    v4f;

#define BN_EPS 1e-4f
#define THR    0.0f

#define BATCH 16
#define CIN   32
#define COUT  32
#define HH    256
#define WW    256
#define TROWS 8          // output rows per block
#define TCOLS 32         // output cols per block
#define CI_STRIDE 40     // padded channel stride (halves): 80B, 16B-aligned, bank-conflict-free
#define ACT_ROWS 10      // TROWS + 2 halo
#define ACT_COLS 34      // TCOLS + 2 halo
#define OUT_CO_STRIDE 33 // padded co stride (floats) for outb: breaks 512B-stride bank collisions

// LDS budget:
//   act : 10*34*40 halves = 27200 B
//   bw  : 9*32*40 halves  = 23040 B   (after act)         total 50240 B
//   outb (alias of base)  : 8*32*33 floats = 33792 B (used after compute barrier)
#define ACT_HALVES (ACT_ROWS*ACT_COLS*CI_STRIDE)
#define BW_HALVES  (9*COUT*CI_STRIDE)

__global__ __launch_bounds__(256)
void tbconv_wmma_kernel(const float* __restrict__ x,
                        const float* __restrict__ gamma,
                        const float* __restrict__ beta,
                        const float* __restrict__ mean,
                        const float* __restrict__ var,
                        const float* __restrict__ w,
                        const float* __restrict__ bias,
                        float* __restrict__ out)
{
    __shared__ __align__(16) unsigned char smem_raw[(ACT_HALVES + BW_HALVES) * 2];
    __shared__ float bn_inv[CIN];
    __shared__ float bn_shift[CIN];

    _Float16* act  = (_Float16*)smem_raw;          // [row][col][ci] stride CI_STRIDE
    _Float16* bw   = act + ACT_HALVES;             // [tap][co][ci]  stride CI_STRIDE
    float*    outb = (float*)smem_raw;             // [r][owl][co]   stride OUT_CO_STRIDE (alias)

    const int tid = threadIdx.x;
    const int bx  = blockIdx.x;
    const int b   = bx / ((HH / TROWS) * (WW / TCOLS));
    const int rem = bx % ((HH / TROWS) * (WW / TCOLS));
    const int oh0 = (rem / (WW / TCOLS)) * TROWS;
    const int ow0 = (rem % (WW / TCOLS)) * TCOLS;

    // Per-channel BN constants into LDS
    if (tid < CIN) {
        float inv = gamma[tid] * __frsqrt_rn(var[tid] + BN_EPS);
        bn_inv[tid]   = inv;
        bn_shift[tid] = beta[tid] - mean[tid] * inv;
    }
    __syncthreads();

    // Weights -> LDS f16, layout bw[tap][co][ci] (coalesced global reads)
    for (int i = tid; i < COUT * CIN * 9; i += 256) {
        int co = i / (CIN * 9);
        int r2 = i % (CIN * 9);
        int ci = r2 / 9;
        int s  = r2 % 9;
        bw[(s * COUT + co) * CI_STRIDE + ci] = (_Float16)w[i];
    }

    // Halo activations: BN -> ternary -> f16 in LDS. Padding is applied to the
    // QUANTIZED tensor (conv pads q with 0), so OOB entries are exactly 0.
    const int ih0 = oh0 - 1, iw0 = ow0 - 1;
    for (int i = tid; i < ACT_ROWS * ACT_COLS * CIN; i += 256) {
        int row = i / (ACT_COLS * CIN);
        int ci  = (i / ACT_COLS) % CIN;
        int lc  = i % ACT_COLS;                    // consecutive tid -> consecutive iw
        int ih  = ih0 + row, iw = iw0 + lc;
        float q = 0.0f;
        if (ih >= 0 && ih < HH && iw >= 0 && iw < WW) {
            float xv = x[(((size_t)b * CIN + ci) * HH + ih) * WW + iw];
            float xh = xv * bn_inv[ci] + bn_shift[ci];
            q = (xh <= -THR) ? -1.0f : ((xh >= THR) ? 1.0f : 0.0f);
        }
        act[(row * ACT_COLS + lc) * CI_STRIDE + ci] = (_Float16)q;
    }
    __syncthreads();

    // ---- Implicit GEMM via v_wmma_f32_16x16x32_f16 ----
    const int lane = tid & 31;
    const int wv   = tid >> 5;        // wave id == output row r (0..7)
    const int m    = lane & 15;       // A: M index / B,D: N index
    const int hi   = lane >> 4;       // lane half

    v8f zero = {0.f, 0.f, 0.f, 0.f, 0.f, 0.f, 0.f, 0.f};
    v8f acc[2][2];                    // [m-half t][cout-half nh]
    acc[0][0] = zero; acc[0][1] = zero; acc[1][0] = zero; acc[1][1] = zero;

    #pragma unroll
    for (int kh = 0; kh < 3; ++kh) {
        #pragma unroll
        for (int kw = 0; kw < 3; ++kw) {
            const int s = kh * 3 + kw;
            // B fragments: lane holds N=m (co = 16*nh+m), elements e -> K(ci)=16*hi+e
            v16h bfr[2];
            #pragma unroll
            for (int nh = 0; nh < 2; ++nh) {
                const _Float16* pb = bw + (s * COUT + (nh * 16 + m)) * CI_STRIDE + 16 * hi;
                v8h blo = *(const v8h*)(pb);
                v8h bhp = *(const v8h*)(pb + 8);
                bfr[nh] = __builtin_shufflevector(blo, bhp,
                          0,1,2,3,4,5,6,7,8,9,10,11,12,13,14,15);
            }
            #pragma unroll
            for (int t = 0; t < 2; ++t) {
                // A fragment: M=m (pixel), element e -> K(ci) = 16*hi + e
                const int lr = wv + kh;
                const int lc = 16 * t + m + kw;
                const _Float16* pa = act + (lr * ACT_COLS + lc) * CI_STRIDE + 16 * hi;
                v8h alo = *(const v8h*)(pa);        // ci = 16*hi .. 16*hi+7
                v8h ahp = *(const v8h*)(pa + 8);    // ci = 16*hi+8 .. 16*hi+15
                v16h a  = __builtin_shufflevector(alo, ahp,
                          0,1,2,3,4,5,6,7,8,9,10,11,12,13,14,15);
                #pragma unroll
                for (int nh = 0; nh < 2; ++nh) {
                    acc[t][nh] = __builtin_amdgcn_wmma_f32_16x16x32_f16(
                        false, a, false, bfr[nh], (short)0, acc[t][nh], false, false);
                }
            }
        }
    }
    __syncthreads();   // done reading act/bw; reuse LDS as outb

    // Bias + ReLU fused here (co is fixed per lane per nh), then scatter to LDS.
    // D layout: VGPR rr, lanes 0-15: M=rr,N=lane; lanes 16-31: M=rr+8,N=lane-16
    float bco[2] = { bias[m], bias[16 + m] };
    #pragma unroll
    for (int t = 0; t < 2; ++t)
        #pragma unroll
        for (int nh = 0; nh < 2; ++nh)
            #pragma unroll
            for (int rr = 0; rr < 8; ++rr) {
                int M   = rr + 8 * hi;
                int owl = 16 * t + M;
                int co  = 16 * nh + m;
                float v = fmaxf(acc[t][nh][rr] + bco[nh], 0.0f);
                outb[(wv * TCOLS + owl) * OUT_CO_STRIDE + co] = v;
            }
    __syncthreads();

    // Wide non-temporal global stores: output is write-once, keep it from
    // evicting the input working set (input ~134MB nearly fits in 192MB L2,
    // halo re-reads between neighbor blocks then hit L2).
    // 2048 units of 4 floats; consecutive tid -> consecutive ow (coalesced b128).
    for (int u = tid; u < (TROWS * TCOLS * COUT) / 4; u += 256) {
        int owl4 = (u & 7) * 4;
        int r    = (u >> 3) & 7;
        int co   = u >> 6;
        v4f v;
        #pragma unroll
        for (int j = 0; j < 4; ++j)
            v[j] = outb[(r * TCOLS + owl4 + j) * OUT_CO_STRIDE + co];
        v4f* dst = (v4f*)(out + (((size_t)b * COUT + co) * HH + (oh0 + r)) * WW + (ow0 + owl4));
        __builtin_nontemporal_store(v, dst);
    }
}

extern "C" void kernel_launch(void* const* d_in, const int* in_sizes, int n_in,
                              void* d_out, int out_size, void* d_ws, size_t ws_size,
                              hipStream_t stream) {
    const float* x     = (const float*)d_in[0];
    const float* gamma = (const float*)d_in[1];
    const float* beta  = (const float*)d_in[2];
    const float* mean  = (const float*)d_in[3];
    const float* var   = (const float*)d_in[4];
    const float* w     = (const float*)d_in[5];
    const float* bias  = (const float*)d_in[6];
    float* out = (float*)d_out;

    const int grid = BATCH * (HH / TROWS) * (WW / TCOLS);   // 16*32*8 = 4096
    tbconv_wmma_kernel<<<grid, 256, 0, stream>>>(x, gamma, beta, mean, var, w, bias, out);
}
